// Net_13984413516193
// MI455X (gfx1250) — compile-verified, matrix operand order checked by reading, not run on hardware
//
#include <hip/hip_runtime.h>

// ---------------------------------------------------------------------------
// Types for WMMA fragments (gfx1250, wave32)
// ---------------------------------------------------------------------------
typedef __attribute__((ext_vector_type(16))) __bf16        v16bf;
typedef __attribute__((ext_vector_type(8)))  float         v8f;
typedef __attribute__((ext_vector_type(4)))  unsigned int  uivec4;

union FragBF { uivec4 q[2]; v16bf v; };

__device__ __forceinline__ unsigned short f2bf(float f) {
    unsigned int u = __float_as_uint(f);
    u += 0x7FFFu + ((u >> 16) & 1u);   // round-to-nearest-even
    return (unsigned short)(u >> 16);
}

// ---------------------------------------------------------------------------
// Elementwise / utility kernels
// ---------------------------------------------------------------------------
__global__ void k_fill_f32(float* p, float v, long long n) {
    long long i = blockIdx.x * (long long)blockDim.x + threadIdx.x;
    if (i < n) p[i] = v;
}
__global__ void k_fill_u32(unsigned int* p, unsigned int v, long long n) {
    long long i = blockIdx.x * (long long)blockDim.x + threadIdx.x;
    if (i < n) p[i] = v;
}
__global__ void k_copy_f32(float* dst, const float* src, long long n) {
    long long i = blockIdx.x * (long long)blockDim.x + threadIdx.x;
    if (i < n) dst[i] = src[i];
}
__global__ void k_bias_relu(float* x, const float* b, long long n, int F) {
    long long i = blockIdx.x * (long long)blockDim.x + threadIdx.x;
    if (i < n) { int f = (int)(i % F); x[i] = fmaxf(x[i] + b[f], 0.f); }
}

// fp32 -> padded bf16 row-major [Mp, Kp]
__global__ void k_cvt_bf16(const float* src, unsigned short* dst,
                           int M, int K, int Mp, int Kp) {
    long long idx = blockIdx.x * (long long)blockDim.x + threadIdx.x;
    long long tot = (long long)Mp * Kp;
    if (idx >= tot) return;
    int i = (int)(idx / Kp), j = (int)(idx % Kp);
    float v = (i < M && j < K) ? src[(size_t)i * K + j] : 0.f;
    dst[idx] = f2bf(v);
}
// fp32 [K,N] -> padded transposed bf16 [Np, Kp]  (dst[n][k] = src[k][n])
__global__ void k_cvt_bf16_T(const float* src, unsigned short* dst,
                             int K, int N, int Np, int Kp) {
    long long idx = blockIdx.x * (long long)blockDim.x + threadIdx.x;
    long long tot = (long long)Np * Kp;
    if (idx >= tot) return;
    int n = (int)(idx / Kp), k = (int)(idx % Kp);
    float v = (n < N && k < K) ? src[(size_t)k * N + n] : 0.f;
    dst[idx] = f2bf(v);
}

// ---------------------------------------------------------------------------
// WMMA GEMM: C[M,N] = A[M,K] * B[K,N]   (A bf16 [Mp,Kp], B bf16 transposed [N,Kp])
// flags: 1 = +bias[n], 2 = relu, 4 = accumulate onto existing C
// One wave computes a 16x64 strip of C (4 N-tiles): each A fragment is loaded
// once and reused by 4 WMMAs -> 4x less A traffic, higher WMMA density.
// Requires N % 64 == 0 (true for all call sites: 768/384/256/128/64).
// ---------------------------------------------------------------------------
__global__ void __launch_bounds__(128)
k_wmma_gemm(const unsigned short* __restrict__ A,
            const unsigned short* __restrict__ B,
            const float* __restrict__ bias,
            float* __restrict__ C,
            int M, int N, int Kp, int Mtiles, int flags) {
    int tileN0 = blockIdx.x * 4;
    int tileM  = blockIdx.y * blockDim.y + threadIdx.y;
    if (tileM >= Mtiles) return;
    int lane = threadIdx.x;
    int half = lane >> 4;
    int r    = lane & 15;

    const unsigned short* Ar = A + (size_t)(tileM  * 16 + r) * Kp;
    const unsigned short* Br = B + (size_t)(tileN0 * 16 + r) * Kp;
    const size_t bstride = (size_t)16 * Kp;   // one N-tile of B rows

    v8f acc[4];
#pragma unroll
    for (int t = 0; t < 4; ++t) acc[t] = v8f{0.f, 0.f, 0.f, 0.f, 0.f, 0.f, 0.f, 0.f};

    for (int kk = 0; kk < Kp; kk += 32) {
        FragBF a;
        a.q[0] = *(const uivec4*)(Ar + kk +      half * 8);
        a.q[1] = *(const uivec4*)(Ar + kk + 16 + half * 8);
        FragBF b[4];
#pragma unroll
        for (int t = 0; t < 4; ++t) {
            const unsigned short* Bt = Br + t * bstride;
            b[t].q[0] = *(const uivec4*)(Bt + kk +      half * 8);
            b[t].q[1] = *(const uivec4*)(Bt + kk + 16 + half * 8);
        }
#pragma unroll
        for (int t = 0; t < 4; ++t) {
            acc[t] = __builtin_amdgcn_wmma_f32_16x16x32_bf16(
                false, a.v, false, b[t].v, (short)0, acc[t], false, false);
        }
    }

#pragma unroll
    for (int t = 0; t < 4; ++t) {
        int n = (tileN0 + t) * 16 + r;
#pragma unroll
        for (int j = 0; j < 8; ++j) {
            int m = tileM * 16 + half * 8 + j;
            if (m < M && n < N) {
                float v = acc[t][j];
                if (flags & 1) v += bias[n];
                if (flags & 4) v += C[(size_t)m * N + n];
                if (flags & 2) v = fmaxf(v, 0.f);
                C[(size_t)m * N + n] = v;
            }
        }
    }
}

// ---------------------------------------------------------------------------
// Graph kernels
// ---------------------------------------------------------------------------
__global__ void k_deg_add(const int* dst, float* deg, int E) {
    int e = blockIdx.x * blockDim.x + threadIdx.x;
    if (e < E) atomicAdd(&deg[dst[e]], 1.f);
}
__global__ void k_rsqrt_deg(float* dis, const float* deg, int n) {
    int i = blockIdx.x * blockDim.x + threadIdx.x;
    if (i < n) dis[i] = rsqrtf(fmaxf(deg[i], 1.f));
}
__global__ void k_edge_norm(const int* src, const int* dst, const float* dis,
                            float* enorm, int E) {
    int e = blockIdx.x * blockDim.x + threadIdx.x;
    if (e < E) enorm[e] = dis[src[e]] * dis[dst[e]];
}
// out[dst[e], f] += w[e] * h[src[e], f]   (w == nullptr -> 1)
__global__ void k_scatter_add(const float* __restrict__ h, const int* src, const int* dst,
                              const float* __restrict__ w, float* out, int E, int F) {
    long long idx = blockIdx.x * (long long)blockDim.x + threadIdx.x;
    long long tot = (long long)E * F;
    if (idx >= tot) return;
    int e = (int)(idx / F), f = (int)(idx % F);
    float ww = w ? w[e] : 1.f;
    atomicAdd(&out[(size_t)dst[e] * F + f], h[(size_t)src[e] * F + f] * ww);
}
__global__ void k_gcn_self(const float* h, const float* dis, float* out, int n, int F) {
    long long idx = blockIdx.x * (long long)blockDim.x + threadIdx.x;
    long long tot = (long long)n * F;
    if (idx >= tot) return;
    int i = (int)(idx / F);
    out[idx] += h[idx] * dis[i] * dis[i];
}
__global__ void k_scale_rows(float* x, const float* deg, int n, int F) {
    long long idx = blockIdx.x * (long long)blockDim.x + threadIdx.x;
    long long tot = (long long)n * F;
    if (idx >= tot) return;
    int i = (int)(idx / F);
    x[idx] /= fmaxf(deg[i], 1.f);
}

// GAT: al[i,hd] = sum_c h[i,hd,c] * a[hd,c]
__global__ void k_gat_alpha(const float* __restrict__ h, const float* __restrict__ a,
                            float* al, int n, int heads, int c) {
    int idx = blockIdx.x * blockDim.x + threadIdx.x;
    if (idx >= n * heads) return;
    int i = idx / heads, hd = idx % heads;
    float s = 0.f;
    const float* hp = h + (size_t)i * heads * c + hd * c;
    const float* ap = a + hd * c;
    for (int j = 0; j < c; ++j) s += hp[j] * ap[j];
    al[idx] = s;
}
__device__ __forceinline__ unsigned int ord_enc(float f) {
    unsigned int u = __float_as_uint(f);
    return (u & 0x80000000u) ? ~u : (u | 0x80000000u);
}
__device__ __forceinline__ float ord_dec(unsigned int u) {
    u = (u & 0x80000000u) ? (u ^ 0x80000000u) : ~u;
    return __uint_as_float(u);
}
// pass1: e = leakyrelu(als[s]+ald[d]); store; atomicMax over dst
__global__ void k_gat_e1(const int* src, const int* dst, const float* als, const float* ald,
                         float* ebuf, unsigned int* menc, int E, int n, int heads) {
    long long idx = blockIdx.x * (long long)blockDim.x + threadIdx.x;
    long long tot = (long long)(E + n) * heads;
    if (idx >= tot) return;
    int e = (int)(idx / heads), hd = (int)(idx % heads);
    int s = (e < E) ? src[e] : (e - E);
    int d = (e < E) ? dst[e] : (e - E);
    float v = als[s * heads + hd] + ald[d * heads + hd];
    v = (v >= 0.f) ? v : 0.2f * v;
    ebuf[idx] = v;
    atomicMax(&menc[d * heads + hd], ord_enc(v));
}
// pass2: ex = exp(e - m[d]); store; atomicAdd den
__global__ void k_gat_e2(const int* dst, const unsigned int* menc,
                         float* ebuf, float* den, int E, int n, int heads) {
    long long idx = blockIdx.x * (long long)blockDim.x + threadIdx.x;
    long long tot = (long long)(E + n) * heads;
    if (idx >= tot) return;
    int e = (int)(idx / heads), hd = (int)(idx % heads);
    int d = (e < E) ? dst[e] : (e - E);
    float ex = expf(ebuf[idx] - ord_dec(menc[d * heads + hd]));
    ebuf[idx] = ex;
    atomicAdd(&den[d * heads + hd], ex);
}
// pass3: out[d,hd,c] += (ex/den[d,hd]) * h[s,hd,c]
__global__ void k_gat_e3(const int* src, const int* dst, const float* __restrict__ ebuf,
                         const float* __restrict__ den, const float* __restrict__ h,
                         float* out, int E, int n, int heads, int c) {
    long long idx = blockIdx.x * (long long)blockDim.x + threadIdx.x;
    int hc = heads * c;
    long long tot = (long long)(E + n) * hc;
    if (idx >= tot) return;
    long long e = idx / hc;
    int rem = (int)(idx % hc);
    int hd = rem / c;
    int s = (e < E) ? src[e] : (int)(e - E);
    int d = (e < E) ? dst[e] : (int)(e - E);
    float alpha = ebuf[e * heads + hd] / den[d * heads + hd];
    atomicAdd(&out[(size_t)d * hc + rem], alpha * h[(size_t)s * hc + rem]);
}

// ---------------------------------------------------------------------------
// BatchNorm (training-mode, biased variance): scale = g*rsqrt(var+eps),
// shift = b - mean*scale  -> apply y = x*scale + shift
// ---------------------------------------------------------------------------
__global__ void __launch_bounds__(256)
k_bn_stats(const float* __restrict__ x, const float* g, const float* b,
           float* scale, float* shift, int n, int F) {
    int f = blockIdx.x;
    int tid = threadIdx.x;
    float s = 0.f, q = 0.f;
    for (int i = tid; i < n; i += 256) {
        float v = x[(size_t)i * F + f];
        s += v; q += v * v;
    }
    __shared__ float ss[256], sq[256];
    ss[tid] = s; sq[tid] = q;
    __syncthreads();
    for (int w = 128; w > 0; w >>= 1) {
        if (tid < w) { ss[tid] += ss[tid + w]; sq[tid] += sq[tid + w]; }
        __syncthreads();
    }
    if (tid == 0) {
        float m = ss[0] / n;
        float var = sq[0] / n - m * m;
        float inv = rsqrtf(var + 1e-5f);
        scale[f] = g[f] * inv;
        shift[f] = b[f] - m * g[f] * inv;
    }
}
__global__ void k_bn_apply(float* x, const float* scale, const float* shift,
                           long long tot, int F) {
    long long i = blockIdx.x * (long long)blockDim.x + threadIdx.x;
    if (i >= tot) return;
    int f = (int)(i % F);
    x[i] = x[i] * scale[f] + shift[f];
}

// ---------------------------------------------------------------------------
// Pool (segment sum into concat layout) + final fc3
// ---------------------------------------------------------------------------
__global__ void k_pool(const float* __restrict__ h, const int* batch, float* pooled,
                       int n, int F, int off, int Ftot) {
    long long idx = blockIdx.x * (long long)blockDim.x + threadIdx.x;
    long long tot = (long long)n * F;
    if (idx >= tot) return;
    int i = (int)(idx / F), f = (int)(idx % F);
    atomicAdd(&pooled[(size_t)batch[i] * Ftot + off + f], h[idx]);
}
__global__ void k_fc3(const float* x, const float* W, const float* b,
                      float* out, int G, int K) {
    int g = blockIdx.x * blockDim.x + threadIdx.x;
    if (g >= G) return;
    float s = b[0];
    for (int j = 0; j < K; ++j) s += x[g * K + j] * W[j];
    out[g] = s;
}

// ---------------------------------------------------------------------------
// Host-side helpers
// ---------------------------------------------------------------------------
static inline unsigned int nblk(long long n, int bs) { return (unsigned int)((n + bs - 1) / bs); }

static void fill_f32(float* p, float v, long long n, hipStream_t s) {
    k_fill_f32<<<nblk(n, 256), 256, 0, s>>>(p, v, n);
}
static void fill_u32(unsigned int* p, unsigned int v, long long n, hipStream_t s) {
    k_fill_u32<<<nblk(n, 256), 256, 0, s>>>(p, v, n);
}

// C[M,N] = A[M,K] @ W[K,N] with epilogue flags (1=bias,2=relu,4=accumulate)
static void run_gemm(const float* A, int M, int K, const float* W, int N,
                     const float* bias, float* C, int flags,
                     unsigned short* Abf, unsigned short* Bbf, hipStream_t s) {
    int Kp = (K + 31) & ~31;
    int Mt = (M + 15) / 16;
    int Mp = Mt * 16;
    long long na = (long long)Mp * Kp;
    k_cvt_bf16<<<nblk(na, 256), 256, 0, s>>>(A, Abf, M, K, Mp, Kp);
    long long nb = (long long)N * Kp;   // N is always a multiple of 64 here
    k_cvt_bf16_T<<<nblk(nb, 256), 256, 0, s>>>(W, Bbf, K, N, N, Kp);
    dim3 blk(32, 4), grd(N / 64, (Mt + 3) / 4);
    k_wmma_gemm<<<grd, blk, 0, s>>>(Abf, Bbf, bias, C, M, N, Kp, Mt, flags);
}

static void run_bn(float* x, const float* g, const float* b,
                   float* scale, float* shift, int n, int F, hipStream_t s) {
    k_bn_stats<<<F, 256, 0, s>>>(x, g, b, scale, shift, n, F);
    long long tot = (long long)n * F;
    k_bn_apply<<<nblk(tot, 256), 256, 0, s>>>(x, scale, shift, tot, F);
}

// ---------------------------------------------------------------------------
// kernel_launch
// ---------------------------------------------------------------------------
extern "C" void kernel_launch(void* const* d_in, const int* in_sizes, int n_in,
                              void* d_out, int out_size, void* d_ws, size_t ws_size,
                              hipStream_t stream) {
    const int NN = 10000, NE = 160000, NG = 64;
    const float* X     = (const float*)d_in[0];
    const int*   src   = (const int*)d_in[1];
    const int*   dst   = src + NE;
    const int*   batch = (const int*)d_in[2];

    const float* P[96];
    for (int i = 3; i < n_in && i < 96; ++i) P[i] = (const float*)d_in[i];

    // ---- workspace carve ----
    char* w = (char*)d_ws;
    size_t off = 0;
    auto carve = [&](size_t bytes) -> void* {
        void* p = w + off;
        off += (bytes + 255) & ~(size_t)255;
        return p;
    };
    float*          bA     = (float*)carve((size_t)NN * 768 * 4);
    float*          bB     = (float*)carve((size_t)NN * 768 * 4);
    float*          bC     = (float*)carve((size_t)NN * 768 * 4);
    unsigned short* Abf    = (unsigned short*)carve((size_t)NN * 768 * 2);
    unsigned short* Bbf    = (unsigned short*)carve((size_t)768 * 768 * 2);
    float*          deg    = (float*)carve((size_t)NN * 4);
    float*          dis    = (float*)carve((size_t)NN * 4);
    float*          enorm  = (float*)carve((size_t)NE * 4);
    float*          ebuf   = (float*)carve((size_t)(NE + NN) * 3 * 4);
    unsigned int*   menc   = (unsigned int*)carve((size_t)NN * 3 * 4);
    float*          den    = (float*)carve((size_t)NN * 3 * 4);
    float*          als    = (float*)carve((size_t)NN * 3 * 4);
    float*          ald    = (float*)carve((size_t)NN * 3 * 4);
    float*          bnsc   = (float*)carve((size_t)768 * 4);
    float*          bnsh   = (float*)carve((size_t)768 * 4);
    float*          pooled = (float*)carve((size_t)NG * 640 * 4);
    float*          fcb1   = (float*)carve((size_t)NG * 256 * 4);
    float*          fcb2   = (float*)carve((size_t)NG * 64 * 4);
    (void)ws_size; (void)in_sizes; (void)out_size;

    // pick two scratch buffers distinct from `cur`
    auto pick2 = [&](const float* cur, float** f1, float** f2) {
        if (cur == bA)      { *f1 = bB; *f2 = bC; }
        else if (cur == bB) { *f1 = bA; *f2 = bC; }
        else if (cur == bC) { *f1 = bA; *f2 = bB; }
        else                { *f1 = bA; *f2 = bB; }   // cur == X
    };

    fill_f32(pooled, 0.f, (long long)NG * 640, stream);

    // =================== GCN branch ===================
    {
        fill_f32(deg, 1.f, NN, stream);                       // self-loops
        k_deg_add<<<nblk(NE, 256), 256, 0, stream>>>(dst, deg, NE);
        k_rsqrt_deg<<<nblk(NN, 256), 256, 0, stream>>>(dis, deg, NN);
        k_edge_norm<<<nblk(NE, 256), 256, 0, stream>>>(src, dst, dis, enorm, NE);

        const float* cur = X; int Fin = 373;
        const int Fo_[3] = {256, 128, 128};
        const int wi_[3] = {3, 7, 11};
        for (int l = 0; l < 3; ++l) {
            int Fo = Fo_[l], wi = wi_[l];
            float *hbuf, *obuf; pick2(cur, &hbuf, &obuf);
            run_gemm(cur, NN, Fin, P[wi], Fo, nullptr, hbuf, 0, Abf, Bbf, stream);
            long long tot = (long long)NN * Fo;
            fill_f32(obuf, 0.f, tot, stream);
            k_scatter_add<<<nblk((long long)NE * Fo, 256), 256, 0, stream>>>(
                hbuf, src, dst, enorm, obuf, NE, Fo);
            k_gcn_self<<<nblk(tot, 256), 256, 0, stream>>>(hbuf, dis, obuf, NN, Fo);
            k_bias_relu<<<nblk(tot, 256), 256, 0, stream>>>(obuf, P[wi + 1], tot, Fo);
            run_bn(obuf, P[wi + 2], P[wi + 3], bnsc, bnsh, NN, Fo, stream);
            cur = obuf; Fin = Fo;
        }
        k_pool<<<nblk((long long)NN * 128, 256), 256, 0, stream>>>(
            cur, batch, pooled, NN, 128, 0, 640);
    }

    // =================== GAT branch ===================
    {
        const float* cur = X; int Fin = 373;
        const int c_[3] = {256, 128, 128};
        const int wb_[3] = {15, 21, 27};
        for (int l = 0; l < 3; ++l) {
            int c = c_[l], Fo = 3 * c, wb = wb_[l];
            float *hbuf, *obuf; pick2(cur, &hbuf, &obuf);
            run_gemm(cur, NN, Fin, P[wb], Fo, nullptr, hbuf, 0, Abf, Bbf, stream);
            k_gat_alpha<<<nblk(NN * 3, 256), 256, 0, stream>>>(hbuf, P[wb + 1], als, NN, 3, c);
            k_gat_alpha<<<nblk(NN * 3, 256), 256, 0, stream>>>(hbuf, P[wb + 2], ald, NN, 3, c);
            fill_u32(menc, 0u, NN * 3, stream);
            fill_f32(den, 0.f, NN * 3, stream);
            long long etot = (long long)(NE + NN) * 3;
            k_gat_e1<<<nblk(etot, 256), 256, 0, stream>>>(src, dst, als, ald, ebuf, menc, NE, NN, 3);
            k_gat_e2<<<nblk(etot, 256), 256, 0, stream>>>(dst, menc, ebuf, den, NE, NN, 3);
            long long tot = (long long)NN * Fo;
            fill_f32(obuf, 0.f, tot, stream);
            k_gat_e3<<<nblk(etot * c, 256), 256, 0, stream>>>(
                src, dst, ebuf, den, hbuf, obuf, NE, NN, 3, c);
            k_bias_relu<<<nblk(tot, 256), 256, 0, stream>>>(obuf, P[wb + 3], tot, Fo);
            run_bn(obuf, P[wb + 4], P[wb + 5], bnsc, bnsh, NN, Fo, stream);
            cur = obuf; Fin = Fo;
        }
        k_pool<<<nblk((long long)NN * 384, 256), 256, 0, stream>>>(
            cur, batch, pooled, NN, 384, 128, 640);
    }

    // =================== GIN branch ===================
    {
        const float* cur = X; int Fin = 373;
        const int dh_[3] = {256, 128, 64};
        const int wb_[3] = {33, 39, 45};
        for (int l = 0; l < 3; ++l) {
            int dh = dh_[l], wb = wb_[l];
            float *agg, *hbuf; pick2(cur, &agg, &hbuf);
            long long ftot = (long long)NN * Fin;
            k_copy_f32<<<nblk(ftot, 256), 256, 0, stream>>>(agg, cur, ftot);
            k_scatter_add<<<nblk((long long)NE * Fin, 256), 256, 0, stream>>>(
                cur, src, dst, nullptr, agg, NE, Fin);
            run_gemm(agg, NN, Fin, P[wb], dh, P[wb + 1], hbuf, 1 | 2, Abf, Bbf, stream);
            // out = relu(h @ W2 + b2), overwrites agg (agg is dead after hbuf)
            run_gemm(hbuf, NN, dh, P[wb + 2], dh, P[wb + 3], agg, 1 | 2, Abf, Bbf, stream);
            run_bn(agg, P[wb + 4], P[wb + 5], bnsc, bnsh, NN, dh, stream);
            cur = agg; Fin = dh;
        }
        k_pool<<<nblk((long long)NN * 64, 256), 256, 0, stream>>>(
            cur, batch, pooled, NN, 64, 512, 640);
    }

    // =================== SAGE branch ===================
    {
        fill_f32(deg, 0.f, NN, stream);
        k_deg_add<<<nblk(NE, 256), 256, 0, stream>>>(dst, deg, NE);

        const float* cur = X; int Fin = 373;
        const int Fo_[3] = {256, 128, 64};
        const int wb_[3] = {51, 56, 61};
        for (int l = 0; l < 3; ++l) {
            int Fo = Fo_[l], wb = wb_[l];
            float *agg, *obuf; pick2(cur, &agg, &obuf);
            long long ftot = (long long)NN * Fin;
            fill_f32(agg, 0.f, ftot, stream);
            k_scatter_add<<<nblk((long long)NE * Fin, 256), 256, 0, stream>>>(
                cur, src, dst, nullptr, agg, NE, Fin);
            k_scale_rows<<<nblk(ftot, 256), 256, 0, stream>>>(agg, deg, NN, Fin);
            run_gemm(agg, NN, Fin, P[wb], Fo, P[wb + 2], obuf, 1, Abf, Bbf, stream);       // agg@Wl + b
            run_gemm(cur, NN, Fin, P[wb + 1], Fo, nullptr, obuf, 4 | 2, Abf, Bbf, stream); // += x@Wr, relu
            run_bn(obuf, P[wb + 3], P[wb + 4], bnsc, bnsh, NN, Fo, stream);
            cur = obuf; Fin = Fo;
        }
        k_pool<<<nblk((long long)NN * 64, 256), 256, 0, stream>>>(
            cur, batch, pooled, NN, 64, 576, 640);
    }

    // =================== MLP head ===================
    run_gemm(pooled, NG, 640, P[66], 256, P[67], fcb1, 1 | 2, Abf, Bbf, stream);
    run_gemm(fcb1,   NG, 256, P[68],  64, P[69], fcb2, 1 | 2, Abf, Bbf, stream);
    k_fc3<<<1, 64, 0, stream>>>(fcb2, P[70], P[71], (float*)d_out, NG, 64);
}
